// BILEAT_62328565400310
// MI455X (gfx1250) — matrix operand order repaired and checked by reading, not compiled, not run on hardware
//
#include <hip/hip_runtime.h>
#include <hip/hip_bf16.h>

typedef unsigned short u16;
typedef __attribute__((ext_vector_type(16))) __bf16 v16bf;
typedef __attribute__((ext_vector_type(8)))  float  v8f;

#define BKP 72   // padded LDS row stride in bf16 elements (144B, 16B-aligned)
#define BK  64   // K-depth staged per LDS tile (2 WMMA k-steps)

__device__ inline u16 f2bf(float f) {
  unsigned u = __float_as_uint(f);
  u += 0x7FFFu + ((u >> 16) & 1u);      // round-to-nearest-even
  return (u16)(u >> 16);
}
__device__ inline float bf2f(u16 h) { return __uint_as_float(((unsigned)h) << 16); }

union FragU { uint4 u[2]; v16bf v; };

// A fragment (16x32 bf16): lane<16: M=lane, K=0..7 & 16..23; lane>=16: M=lane-16, K=8..15 & 24..31
__device__ inline v16bf frag_ld_a(const u16* lds, int row0, int ksub, int lane) {
  int l = lane & 15, hi = lane >> 4;
  const u16* p = lds + (size_t)(row0 + l) * BKP + ksub + hi * 8;
  FragU f;
  f.u[0] = *(const uint4*)(p);
  f.u[1] = *(const uint4*)(p + 16);
  return f.v;
}
// B fragment (32x16 bf16): lane<16: N=lane, K=0..7 & 16..23; lane>=16: N=lane-16, K=8..15 & 24..31
__device__ inline v16bf frag_ld_b(const u16* lds, int col0, int ksub, int lane) {
  int l = lane & 15, hi = lane >> 4;
  const u16* p = lds + (size_t)(col0 + l) * BKP + ksub + hi * 8;
  FragU f;
  f.u[0] = *(const uint4*)(p);
  f.u[1] = *(const uint4*)(p + 16);
  return f.v;
}

// C[M,N] = A[M,K] * B[K,N], batched over blockIdx.z.
// ABF/BBF: source is bf16 (else fp32, converted while staging to LDS)
// BNK: Bsrc stored [N,K] row-major (already B^T)
// EPI: 0 = bf16 store + bias ; 1 = bf16 store ; 2 = fp32 store with (NEG*eye, dist) epilogue
template<int ABF, int BBF, int BNK, int EPI>
__global__ __launch_bounds__(256) void wmma_gemm(
    const void* Asrc, long long sA, int lda,
    const void* Bsrc, long long sB, int ldb,
    void* Cdst, long long sC, int ldc,
    const float* bias, int K)
{
  __shared__ __align__(16) u16 lds_a[128 * BKP];
  __shared__ __align__(16) u16 lds_b[128 * BKP];

  const int tid  = threadIdx.x;
  const int lane = tid & 31;
  const int wid  = tid >> 5;
  const int wm   = wid & 3;          // 4 waves along M, 32 rows each
  const int wn   = wid >> 2;         // 2 waves along N, 64 cols each
  const int bm   = blockIdx.y * 128;
  const int bn   = blockIdx.x * 128;
  const size_t aOff = (size_t)blockIdx.z * (size_t)sA;
  const size_t bOff = (size_t)blockIdx.z * (size_t)sB;
  const size_t cOff = (size_t)blockIdx.z * (size_t)sC;

  v8f acc[2][4];
  #pragma unroll
  for (int im = 0; im < 2; ++im)
    #pragma unroll
    for (int in = 0; in < 4; ++in) {
      v8f z = {0.f,0.f,0.f,0.f,0.f,0.f,0.f,0.f};
      acc[im][in] = z;
    }

  for (int k0 = 0; k0 < K; k0 += BK) {
    // ---- stage A tile (128 x 64) -> LDS, 8 elements/thread/iter, b128 stores ----
    if (ABF) {
      #pragma unroll
      for (int i = 0; i < 4; ++i) {
        int t = tid + i * 256;                 // 0..1023
        int r = t >> 3, kq = (t & 7) * 8;
        const u16* src = (const u16*)Asrc + aOff + (size_t)(bm + r) * lda + (k0 + kq);
        *(uint4*)&lds_a[r * BKP + kq] = *(const uint4*)src;
      }
    } else {
      #pragma unroll
      for (int i = 0; i < 4; ++i) {
        int t = tid + i * 256;
        int r = t >> 3, kq = (t & 7) * 8;
        const float* src = (const float*)Asrc + aOff + (size_t)(bm + r) * lda + (k0 + kq);
        float4 f0 = *(const float4*)(src);
        float4 f1 = *(const float4*)(src + 4);
        u16 h[8] = { f2bf(f0.x), f2bf(f0.y), f2bf(f0.z), f2bf(f0.w),
                     f2bf(f1.x), f2bf(f1.y), f2bf(f1.z), f2bf(f1.w) };
        *(uint4*)&lds_a[r * BKP + kq] = *(const uint4*)h;
      }
    }
    // ---- stage B tile as [n][k] ----
    if (BNK) {
      // Bsrc is [N,K]: same pattern as A
      if (BBF) {
        #pragma unroll
        for (int i = 0; i < 4; ++i) {
          int t = tid + i * 256;
          int n = t >> 3, kq = (t & 7) * 8;
          const u16* src = (const u16*)Bsrc + bOff + (size_t)(bn + n) * ldb + (k0 + kq);
          *(uint4*)&lds_b[n * BKP + kq] = *(const uint4*)src;
        }
      } else {
        #pragma unroll
        for (int i = 0; i < 4; ++i) {
          int t = tid + i * 256;
          int n = t >> 3, kq = (t & 7) * 8;
          const float* src = (const float*)Bsrc + bOff + (size_t)(bn + n) * ldb + (k0 + kq);
          float4 f0 = *(const float4*)(src);
          float4 f1 = *(const float4*)(src + 4);
          u16 h[8] = { f2bf(f0.x), f2bf(f0.y), f2bf(f0.z), f2bf(f0.w),
                       f2bf(f1.x), f2bf(f1.y), f2bf(f1.z), f2bf(f1.w) };
          *(uint4*)&lds_b[n * BKP + kq] = *(const uint4*)h;
        }
      }
    } else {
      // Bsrc is [K,N]: vector-load along n, scatter-transpose into LDS
      if (BBF) {
        #pragma unroll
        for (int i = 0; i < 4; ++i) {
          int t = tid + i * 256;                // 0..1023
          int k = t >> 4, n8 = (t & 15) * 8;
          const u16* src = (const u16*)Bsrc + bOff + (size_t)(k0 + k) * ldb + (bn + n8);
          uint4 raw = *(const uint4*)src;
          const u16* hp = (const u16*)&raw;
          #pragma unroll
          for (int j = 0; j < 8; ++j) lds_b[(n8 + j) * BKP + k] = hp[j];
        }
      } else {
        #pragma unroll
        for (int i = 0; i < 8; ++i) {
          int t = tid + i * 256;                // 0..2047
          int k = t >> 5, n4 = (t & 31) * 4;
          const float* src = (const float*)Bsrc + bOff + (size_t)(k0 + k) * ldb + (bn + n4);
          float4 f = *(const float4*)src;
          lds_b[(n4 + 0) * BKP + k] = f2bf(f.x);
          lds_b[(n4 + 1) * BKP + k] = f2bf(f.y);
          lds_b[(n4 + 2) * BKP + k] = f2bf(f.z);
          lds_b[(n4 + 3) * BKP + k] = f2bf(f.w);
        }
      }
    }
    // prefetch next A tile (global_prefetch_b8)
    if (k0 + BK < K) {
      int r = tid >> 1, kq = (tid & 1) * 32;
      size_t gi = aOff + (size_t)(bm + r) * lda + (size_t)(k0 + BK + kq);
      if (ABF) __builtin_prefetch(((const u16*)Asrc) + gi, 0, 1);
      else     __builtin_prefetch(((const float*)Asrc) + gi, 0, 1);
    }
    __syncthreads();

    // ---- two WMMA k-steps per staged tile: 16 v_wmma per iteration ----
    #pragma unroll
    for (int ks = 0; ks < 2; ++ks) {
      v16bf af[2], bfv[4];
      #pragma unroll
      for (int im = 0; im < 2; ++im) af[im]  = frag_ld_a(lds_a, wm * 32 + im * 16, ks * 32, lane);
      #pragma unroll
      for (int in = 0; in < 4; ++in) bfv[in] = frag_ld_b(lds_b, wn * 64 + in * 16, ks * 32, lane);
      #pragma unroll
      for (int im = 0; im < 2; ++im)
        #pragma unroll
        for (int in = 0; in < 4; ++in)
          acc[im][in] = __builtin_amdgcn_wmma_f32_16x16x32_bf16(
              false, af[im], false, bfv[in], (short)0, acc[im][in], false, false);
    }
    __syncthreads();
  }

  // ---- epilogue ----
  const int col_l = lane & 15;
  const int rofs  = (lane >> 4) * 8;
  #pragma unroll
  for (int im = 0; im < 2; ++im) {
    #pragma unroll
    for (int in = 0; in < 4; ++in) {
      int m_base = bm + wm * 32 + im * 16;
      int n_base = bn + wn * 64 + in * 16;
      int col = n_base + col_l;
      v8f a = acc[im][in];
      #pragma unroll
      for (int r = 0; r < 8; ++r) {
        int row = m_base + r + rofs;
        float val = a[r];
        size_t ci = cOff + (size_t)row * ldc + col;
        if (EPI == 0) {
          ((u16*)Cdst)[ci] = f2bf(val + bias[col]);
        } else if (EPI == 1) {
          ((u16*)Cdst)[ci] = f2bf(val);
        } else {
          // A = (hW.h_op^T + NEG*eye) * dist
          float diag = (row == col) ? 1.f : 0.f;
          val += diag * -1e18f;
          float d = 1.f / (fabsf((float)(row - col)) + diag);
          ((float*)Cdst)[ci] = val * d;
        }
      }
    }
  }
}

__global__ void init_kernel(float* acc) {
  if (threadIdx.x < 8) acc[threadIdx.x] = 0.f;
}

// per-token: o_ae/o_op heads (NN=3), softmax prob sums, margin losses, mask count
__global__ __launch_bounds__(256) void head_kernel(
    const u16* h_ae, const u16* h_op,
    const float* w_ae, const float* b_ae,
    const float* w_op, const float* b_op,
    const int* amask, const int* lab_n, const int* lab_o,
    float* pae_s, float* pop_s, float* acc)
{
  int lane = threadIdx.x & 31, wid = threadIdx.x >> 5;
  int tok = blockIdx.x * 8 + wid;
  float m = (amask[tok] == 1) ? 1.f : 0.f;
  const u16*   hs[2] = { h_ae, h_op };
  const float* ws[2] = { w_ae, w_op };
  const float* bs[2] = { b_ae, b_op };
  const int*   ls[2] = { lab_n, lab_o };
  float*       ps[2] = { pae_s, pop_s };
  #pragma unroll
  for (int p = 0; p < 2; ++p) {
    float o0 = 0.f, o1 = 0.f, o2 = 0.f;
    const u16* h = hs[p] + (size_t)tok * 768;
    for (int j = 0; j < 24; ++j) {
      int idx = lane + j * 32;
      float v = fmaxf(bf2f(h[idx]), 0.f);   // relu
      const float* w = ws[p] + idx * 3;
      o0 += v * w[0]; o1 += v * w[1]; o2 += v * w[2];
    }
    for (int off = 16; off; off >>= 1) {
      o0 += __shfl_xor(o0, off, 32);
      o1 += __shfl_xor(o1, off, 32);
      o2 += __shfl_xor(o2, off, 32);
    }
    if (lane == 0) {
      float o[3] = { o0 + bs[p][0], o1 + bs[p][1], o2 + bs[p][2] };
      float mx = fmaxf(o[0], fmaxf(o[1], o[2]));
      float e0 = __expf(o[0] - mx), e1 = __expf(o[1] - mx), e2 = __expf(o[2] - mx);
      float s = e0 + e1 + e2;
      ps[p][tok] = (e1 + e2) / s;
      int y = ls[p][tok];
      float per = 0.f;
      for (int j = 0; j < 3; ++j)
        if (j != y) per += fmaxf(0.f, 1.f - o[y] + o[j]);
      atomicAdd(&acc[p], (per * (1.f / 3.f)) * m);
      if (p == 0) atomicAdd(&acc[3], m);
    }
  }
}

// row/col softmax of (A * p), output bf16 probabilities
__global__ __launch_bounds__(256) void softmax_kernel(
    const float* A, const float* psc, u16* P, int transposed)
{
  __shared__ float red[256];
  int r = blockIdx.x;
  int b = r >> 10, s = r & 1023;
  const float* Ab = A + ((size_t)b << 20);
  const float* pb = psc + (b << 10);
  int tid = threadIdx.x;
  float v[4];
  float mx = -3.4e38f;
  #pragma unroll
  for (int i = 0; i < 4; ++i) {
    int t = tid + i * 256;
    float a = transposed ? Ab[(size_t)t * 1024 + s] : Ab[(size_t)s * 1024 + t];
    float x = a * pb[t];
    v[i] = x; mx = fmaxf(mx, x);
  }
  red[tid] = mx; __syncthreads();
  for (int o = 128; o > 0; o >>= 1) { if (tid < o) red[tid] = fmaxf(red[tid], red[tid + o]); __syncthreads(); }
  mx = red[0]; __syncthreads();
  float sum = 0.f;
  #pragma unroll
  for (int i = 0; i < 4; ++i) { v[i] = __expf(v[i] - mx); sum += v[i]; }
  red[tid] = sum; __syncthreads();
  for (int o = 128; o > 0; o >>= 1) { if (tid < o) red[tid] += red[tid + o]; __syncthreads(); }
  float inv = 1.f / red[0];
  u16* Pr = P + ((size_t)r << 10);
  #pragma unroll
  for (int i = 0; i < 4; ++i) { int t = tid + i * 256; Pr[t] = f2bf(v[i] * inv); }
}

// o_prime = concat(h_ae, ae', h_op, op') @ w_fc + b_fc ; margin-3 hinge loss
__global__ __launch_bounds__(256) void final_kernel(
    const u16* h_ae, const u16* ae_p, const u16* h_op, const u16* op_p,
    const float* w_fc, const float* b_fc,
    const int* labels, const int* amask,
    float* out, float* acc)
{
  int lane = threadIdx.x & 31, wid = threadIdx.x >> 5;
  int tok = blockIdx.x * 8 + wid;
  const u16* seg[4] = { h_ae + (size_t)tok * 768, ae_p + (size_t)tok * 768,
                        h_op + (size_t)tok * 768, op_p + (size_t)tok * 768 };
  float o[7] = {0,0,0,0,0,0,0};
  #pragma unroll
  for (int g = 0; g < 4; ++g) {
    const u16* h = seg[g];
    for (int j = 0; j < 24; ++j) {
      int d = lane + j * 32;
      float v = bf2f(h[d]);
      const float* w = w_fc + (size_t)(g * 768 + d) * 7;
      #pragma unroll
      for (int c = 0; c < 7; ++c) o[c] += v * w[c];
    }
  }
  #pragma unroll
  for (int c = 0; c < 7; ++c)
    for (int off = 16; off; off >>= 1) o[c] += __shfl_xor(o[c], off, 32);
  if (lane == 0) {
    float m = (amask[tok] == 1) ? 1.f : 0.f;
    #pragma unroll
    for (int c = 0; c < 7; ++c) o[c] += b_fc[c];
    float* op = out + (size_t)tok * 7;
    #pragma unroll
    for (int c = 0; c < 7; ++c) op[c] = o[c];
    int y = labels[tok];
    float per = 0.f;
    for (int j = 0; j < 7; ++j)
      if (j != y) per += fmaxf(0.f, 3.f - o[y] + o[j]);
    atomicAdd(&acc[2], (per * (1.f / 7.f)) * m);
  }
}

__global__ void combine_kernel(const float* acc, float* out) {
  float m = fmaxf(acc[3], 1.f);
  out[0] = (0.1f * acc[0] + 0.1f * acc[1] + acc[2]) / m;
}

extern "C" void kernel_launch(void* const* d_in, const int* in_sizes, int n_in,
                              void* d_out, int out_size, void* d_ws, size_t ws_size,
                              hipStream_t stream) {
  (void)in_sizes; (void)n_in; (void)out_size; (void)ws_size;
  const float* x     = (const float*)d_in[0];
  const float* w_ae1 = (const float*)d_in[1];
  const float* b_ae1 = (const float*)d_in[2];
  const float* w_op1 = (const float*)d_in[3];
  const float* b_op1 = (const float*)d_in[4];
  const float* w_ae  = (const float*)d_in[5];
  const float* b_ae  = (const float*)d_in[6];
  const float* w_op  = (const float*)d_in[7];
  const float* b_op  = (const float*)d_in[8];
  const float* W     = (const float*)d_in[9];
  const float* w_fc  = (const float*)d_in[10];
  const float* b_fc  = (const float*)d_in[11];
  const int* amask   = (const int*)d_in[12];
  const int* labels  = (const int*)d_in[13];
  const int* lab_n   = (const int*)d_in[14];
  const int* lab_o   = (const int*)d_in[15];
  float* out = (float*)d_out;

  const size_t SZ_H = (size_t)32768 * 768;          // [B*S, 2D]
  const size_t SZ_P = (size_t)32 * 1024 * 1024;     // [B, S, S]
  u16* h_ae = (u16*)d_ws;
  u16* h_op = h_ae + SZ_H;
  u16* hW   = h_op + SZ_H;
  u16* op_p = hW   + SZ_H;
  u16* ae_p = op_p + SZ_H;
  u16* Pop  = ae_p + SZ_H;
  u16* Pae  = Pop  + SZ_P;
  float* Asc   = (float*)(Pae + SZ_P);
  float* pae_s = Asc + SZ_P;
  float* pop_s = pae_s + 32768;
  float* acc   = pop_s + 32768;

  dim3 blk(256);
  init_kernel<<<1, 32, 0, stream>>>(acc);

  // h_ae = x @ w_ae1 + b_ae1 ; h_op = x @ w_op1 + b_op1     [32768x768]x[768x768]
  wmma_gemm<0,0,0,0><<<dim3(6, 256, 1), blk, 0, stream>>>(x, 0, 768, w_ae1, 0, 768,
                                                          h_ae, 0, 768, b_ae1, 768);
  wmma_gemm<0,0,0,0><<<dim3(6, 256, 1), blk, 0, stream>>>(x, 0, 768, w_op1, 0, 768,
                                                          h_op, 0, 768, b_op1, 768);

  // heads, prob sums, first two margin losses, mask count
  head_kernel<<<4096, blk, 0, stream>>>(h_ae, h_op, w_ae, b_ae, w_op, b_op,
                                        amask, lab_n, lab_o, pae_s, pop_s, acc);

  // hW = h_ae @ W
  wmma_gemm<1,0,0,1><<<dim3(6, 256, 1), blk, 0, stream>>>(h_ae, 0, 768, W, 0, 768,
                                                          hW, 0, 768, nullptr, 768);

  // A = (hW @ h_op^T + NEG*eye) * dist   (batched, fp32 out)
  wmma_gemm<1,1,1,2><<<dim3(8, 8, 32), blk, 0, stream>>>(hW, 786432, 768, h_op, 786432, 768,
                                                         Asc, 1048576, 1024, nullptr, 768);

  // Pop = softmax_t(A[s,t]*p_op[t]) ; Pae = softmax_t(A[t,s]*p_ae[t])
  softmax_kernel<<<32768, blk, 0, stream>>>(Asc, pop_s, Pop, 0);
  softmax_kernel<<<32768, blk, 0, stream>>>(Asc, pae_s, Pae, 1);

  // op' = Pop @ h_op ; ae' = Pae @ h_ae   (batched [1024x1024]x[1024x768])
  wmma_gemm<1,1,0,1><<<dim3(6, 8, 32), blk, 0, stream>>>(Pop, 1048576, 1024, h_op, 786432, 768,
                                                         op_p, 786432, 768, nullptr, 1024);
  wmma_gemm<1,1,0,1><<<dim3(6, 8, 32), blk, 0, stream>>>(Pae, 1048576, 1024, h_ae, 786432, 768,
                                                         ae_p, 786432, 768, nullptr, 1024);

  // o_prime + final margin loss ; then combine scalar loss
  final_kernel<<<4096, blk, 0, stream>>>(h_ae, ae_p, h_op, op_p, w_fc, b_fc,
                                         labels, amask, out + 1, acc);
  combine_kernel<<<1, 1, 0, stream>>>(acc, out);
}
